// TrajectoryGenerator_31808527794560
// MI455X (gfx1250) — matrix operand order repaired and checked by reading, not compiled, not run on hardware
//
#include <hip/hip_runtime.h>
#include <hip/hip_bf16.h>
#include <math.h>

// ---------------------------------------------------------------------------
// Types / WMMA helpers (gfx1250 wave32, V_WMMA_F32_16X16X32_F16)
// ---------------------------------------------------------------------------
typedef _Float16 half_t;
typedef __attribute__((ext_vector_type(16))) _Float16 v16h;
typedef __attribute__((ext_vector_type(8)))  float    v8f;

struct alignas(16) h8v { half_t d[8]; };

#define GF_RELU 1

__device__ __forceinline__ v16h make_frag(const half_t* p0, const half_t* p1) {
  h8v lo = *(const h8v*)p0;
  h8v hi = *(const h8v*)(p1);
  v16h f;
#pragma unroll
  for (int e = 0; e < 8; e++) { f[e] = lo.d[e]; f[8 + e] = hi.d[e]; }
  return f;
}

__device__ __forceinline__ v8f wmma_f16(v16h a, v16h b, v8f c) {
  return __builtin_amdgcn_wmma_f32_16x16x32_f16(false, a, false, b, (short)0, c,
                                                false, false);
}

// Software-pipelined K-loop: acc[0..3] += A[16 x 32*ktC] @ B[tiles ktS..]
__device__ __forceinline__ void gemm_acc4(v8f acc[4], const half_t* __restrict__ A,
                                          int lda, const half_t* __restrict__ Bp,
                                          int KTtot, int ktS, int ktC, int rowBase,
                                          int colBase, int lane) {
  const int r  = lane & 15;
  const int hf = lane >> 4;
  const half_t* pa = A + (size_t)(rowBase + r) * lda + 8 * hf;
  const half_t* pb0 = Bp + ((((size_t)(colBase >> 4) + 0) * KTtot + ktS) * 32 + lane) * 16;
  const half_t* pb1 = Bp + ((((size_t)(colBase >> 4) + 1) * KTtot + ktS) * 32 + lane) * 16;
  const half_t* pb2 = Bp + ((((size_t)(colBase >> 4) + 2) * KTtot + ktS) * 32 + lane) * 16;
  const half_t* pb3 = Bp + ((((size_t)(colBase >> 4) + 3) * KTtot + ktS) * 32 + lane) * 16;

  v16h a_c  = make_frag(pa, pa + 16);
  v16h b0_c = make_frag(pb0, pb0 + 8);
  v16h b1_c = make_frag(pb1, pb1 + 8);
  v16h b2_c = make_frag(pb2, pb2 + 8);
  v16h b3_c = make_frag(pb3, pb3 + 8);

  for (int q = 0; q < ktC - 1; q++) {
    pa += 32; pb0 += 512; pb1 += 512; pb2 += 512; pb3 += 512;
    v16h a_n  = make_frag(pa, pa + 16);
    v16h b0_n = make_frag(pb0, pb0 + 8);
    v16h b1_n = make_frag(pb1, pb1 + 8);
    v16h b2_n = make_frag(pb2, pb2 + 8);
    v16h b3_n = make_frag(pb3, pb3 + 8);
    acc[0] = wmma_f16(a_c, b0_c, acc[0]);
    acc[1] = wmma_f16(a_c, b1_c, acc[1]);
    acc[2] = wmma_f16(a_c, b2_c, acc[2]);
    acc[3] = wmma_f16(a_c, b3_c, acc[3]);
    a_c = a_n; b0_c = b0_n; b1_c = b1_n; b2_c = b2_n; b3_c = b3_n;
  }
  acc[0] = wmma_f16(a_c, b0_c, acc[0]);
  acc[1] = wmma_f16(a_c, b1_c, acc[1]);
  acc[2] = wmma_f16(a_c, b2_c, acc[2]);
  acc[3] = wmma_f16(a_c, b3_c, acc[3]);
}

// ---------------------------------------------------------------------------
// Pack a row-major f32 weight [K,N] into per-lane-contiguous B fragments.
// B layout (ISA 7.12.2): lane = 16*half + col, element e -> K = (e&7) + 8*half + 16*(e>>3).
// Packed index: (((nt*KT + kt)*32 + lane)*16 + e)
// ---------------------------------------------------------------------------
__global__ void pack_b_kernel(const float* __restrict__ W, half_t* __restrict__ out,
                              int K, int N) {
  int idx = blockIdx.x * blockDim.x + threadIdx.x;
  if (idx >= K * N) return;
  int e    = idx & 15;
  int lane = (idx >> 4) & 31;
  int frag = idx >> 9;
  int KT   = K >> 5;
  int kt   = frag % KT;
  int nt   = frag / KT;
  int k    = kt * 32 + (e & 7) + 8 * (lane >> 4) + 16 * (e >> 3);
  int col  = nt * 16 + (lane & 15);
  out[idx] = (half_t)W[(size_t)k * N + col];
}

// ---------------------------------------------------------------------------
// Dual-input WMMA GEMM:
//   X = A1 @ B1[ktS1..] (+ A2 @ B2[ktS2..]) + bias1 (+ bias2) (+ rowBias[row&127])
//   C = act(X) (f32, optional), O16 = f16(act(X)) (optional)
// block = 256 threads = 8 waves; wave -> 16 rows x 64 cols; grid = (N/64, M/128)
// ---------------------------------------------------------------------------
__global__ __launch_bounds__(256) void gemm16_kernel(
    const half_t* __restrict__ A1, int lda1, const half_t* __restrict__ Bp1,
    int KTtot1, int ktS1, int ktC1,
    const half_t* __restrict__ A2, int lda2, const half_t* __restrict__ Bp2,
    int KTtot2, int ktS2, int ktC2,
    const float* __restrict__ bias1, const float* __restrict__ bias2,
    const float* __restrict__ rowBias, int rbLd,
    float* __restrict__ C, half_t* __restrict__ O16,
    int N, int M, int flags) {
  const int lane    = threadIdx.x & 31;
  const int wave    = threadIdx.x >> 5;
  const int rowBase = (blockIdx.y * 8 + wave) * 16;
  if (rowBase >= M) return;  // wave-uniform: EXEC stays all-1 for WMMA
  const int colBase = blockIdx.x * 64;
  const int r  = lane & 15;
  const int hf = lane >> 4;

  v8f acc[4];
  {
    v8f z = {};
#pragma unroll
    for (int t = 0; t < 4; t++) acc[t] = z;
  }

  gemm_acc4(acc, A1, lda1, Bp1, KTtot1, ktS1, ktC1, rowBase, colBase, lane);
  if (A2) gemm_acc4(acc, A2, lda2, Bp2, KTtot2, ktS2, ktC2, rowBase, colBase, lane);

#pragma unroll
  for (int t = 0; t < 4; t++) {
    const int col = colBase + 16 * t + r;
    float bb = 0.f;
    if (bias1) bb += bias1[col];
    if (bias2) bb += bias2[col];
#pragma unroll
    for (int v = 0; v < 8; v++) {
      const int row = rowBase + v + 8 * hf;
      float val = acc[t][v] + bb;
      if (rowBias) val += rowBias[(size_t)(row & 127) * rbLd + col];
      if (flags & GF_RELU) val = val > 0.f ? val : 0.f;
      if (C)   C[(size_t)row * N + col]   = val;
      if (O16) O16[(size_t)row * N + col] = (half_t)val;
    }
  }
}

// ---------------------------------------------------------------------------
// Fused pool layer 1 (embedding computed in registers):
//   pre1[i*128+j, c] = relu( emb(pos[j]-pos[i]) @ W1[:64, c] + b1[c] + hW1[j, c] )
// where emb(rel)[k] = rel0*eW[0,k] + rel1*eW[1,k] + eb[k].
// A fragments are synthesized per-lane with packed f16 FMAs -- no A memory.
// block = 256 threads = 8 waves; wave -> 16 rows x 64 cols; grid = (8, 128)
// ---------------------------------------------------------------------------
__global__ __launch_bounds__(256) void pool_l1_kernel(
    const float* __restrict__ pos, const float* __restrict__ eW,
    const float* __restrict__ eb, const half_t* __restrict__ Bp, int KTtot,
    const float* __restrict__ b1, const float* __restrict__ hW1,
    half_t* __restrict__ pre1) {
  const int lane    = threadIdx.x & 31;
  const int wave    = threadIdx.x >> 5;
  const int rowBase = (blockIdx.y * 8 + wave) * 16;  // M = 16384
  const int colBase = blockIdx.x * 64;               // N = 512
  const int r  = lane & 15;
  const int hf = lane >> 4;

  // per-lane embedding weights for this lane's 32 k-values (2 k-tiles)
  v16h vw0[2], vw1[2], vwb[2];
#pragma unroll
  for (int q = 0; q < 2; q++)
#pragma unroll
    for (int e = 0; e < 16; e++) {
      int k = (e & 7) + 8 * hf + 16 * (e >> 3) + 32 * q;  // A-fragment K map
      vw0[q][e] = (half_t)eW[k];
      vw1[q][e] = (half_t)eW[64 + k];
      vwb[q][e] = (half_t)eb[k];
    }

  // this lane's A-row: row = i*128 + j
  const int row = rowBase + r;
  const int i = row >> 7, j = row & 127;
  const half_t rel0 = (half_t)(pos[j * 2]     - pos[i * 2]);
  const half_t rel1 = (half_t)(pos[j * 2 + 1] - pos[i * 2 + 1]);
  v16h s0, s1;
#pragma unroll
  for (int e = 0; e < 16; e++) { s0[e] = rel0; s1[e] = rel1; }

  v8f acc[4];
  {
    v8f z = {};
#pragma unroll
    for (int t = 0; t < 4; t++) acc[t] = z;
  }
#pragma unroll
  for (int q = 0; q < 2; q++) {
    v16h a = s0 * vw0[q] + s1 * vw1[q] + vwb[q];  // v_pk_fma_f16
#pragma unroll
    for (int t = 0; t < 4; t++) {
      const half_t* pb =
          Bp + ((((size_t)(colBase >> 4) + t) * KTtot + q) * 32 + lane) * 16;
      v16h b = make_frag(pb, pb + 8);
      acc[t] = wmma_f16(a, b, acc[t]);
    }
  }

#pragma unroll
  for (int t = 0; t < 4; t++) {
    const int col = colBase + 16 * t + r;
    const float bb = b1[col];
#pragma unroll
    for (int v = 0; v < 8; v++) {
      const int rw = rowBase + v + 8 * hf;
      float val = acc[t][v] + bb + hW1[(size_t)(rw & 127) * 512 + col];
      val = val > 0.f ? val : 0.f;
      pre1[(size_t)rw * 512 + col] = (half_t)val;
    }
  }
}

// ---------------------------------------------------------------------------
// Fused pool layer 2 + max-over-j, async-LDS-staged B:
//   poolh[i, c] = max_{j<128} relu( pre1[i*128+j, :512] @ W2[:, c] + b2[c] )
// block = 128 threads = 4 waves; WG covers (i, 64 cols).
//   - B block (64 KB packed) DMA'd to LDS via GLOBAL_LOAD_ASYNC_TO_LDS_B128
//   - waves split j-tiles (wave w -> jt {w, w+4}); A fetched once per WG
//   - cross-wave max via small LDS reduction
// grid = (1024/64, 128 i's); dynamic LDS = 65536 + 1024 bytes
// ---------------------------------------------------------------------------
__global__ __launch_bounds__(128) void pool_l2max_kernel(
    const half_t* __restrict__ pre1, const half_t* __restrict__ Bp,
    const float* __restrict__ b2, half_t* __restrict__ poolh) {
  extern __shared__ char smem[];
  half_t* Bs   = (half_t*)smem;            // 64 frag-groups * 512 halves = 64 KB
  float*  maxS = (float*)(smem + 65536);   // [4][64]

  const int tid  = threadIdx.x;
  const int lane = tid & 31;
  const int wave = tid >> 5;
  const int i    = blockIdx.y;
  const int cb   = blockIdx.x;             // 64-col block
  const int r    = lane & 15;
  const int hf   = lane >> 4;

  // ---- async B block copy: contiguous 64 KB (frags [cb*64, cb*64+64)) ----
  {
    const char* src = (const char*)Bp + (size_t)cb * 65536;
    // low 32 bits of flat shared address = workgroup-relative LDS byte offset
    const unsigned dstBase = (unsigned)(size_t)(void*)Bs;
    for (int u = tid; u < 4096; u += 128) {  // 4096 x 16B units, 32 per lane
      asm volatile("global_load_async_to_lds_b128 %0, %1, off"
                   :: "v"(dstBase + u * 16), "v"(src + (size_t)u * 16)
                   : "memory");
    }
    asm volatile("s_wait_asynccnt 0" ::: "memory");
  }
  __syncthreads();

  float bb[4];
#pragma unroll
  for (int t = 0; t < 4; t++) bb[t] = b2[cb * 64 + t * 16 + r];

  float runmax[4] = {0.f, 0.f, 0.f, 0.f};  // relu >= 0 -> 0 is valid identity

  for (int jj = 0; jj < 2; jj++) {
    const int jt = wave + 4 * jj;          // this wave's j-tile
    v8f acc[4];
    {
      v8f z = {};
#pragma unroll
      for (int t = 0; t < 4; t++) acc[t] = z;
    }
    const half_t* pa = pre1 + ((size_t)i * 128 + jt * 16 + r) * 512 + 8 * hf;
    v16h a_c = make_frag(pa, pa + 16);
    for (int kt = 0; kt < 15; kt++) {
      pa += 32;
      v16h a_n = make_frag(pa, pa + 16);   // global prefetch of next k-tile
#pragma unroll
      for (int t = 0; t < 4; t++) {
        const half_t* pb = Bs + (((size_t)t * 16 + kt) * 32 + lane) * 16;
        v16h b = make_frag(pb, pb + 8);    // ds_load_b128 x2
        acc[t] = wmma_f16(a_c, b, acc[t]);
      }
      a_c = a_n;
    }
#pragma unroll
    for (int t = 0; t < 4; t++) {          // kt = 15 tail
      const half_t* pb = Bs + (((size_t)t * 16 + 15) * 32 + lane) * 16;
      v16h b = make_frag(pb, pb + 8);
      acc[t] = wmma_f16(a_c, b, acc[t]);
    }
#pragma unroll
    for (int t = 0; t < 4; t++)
#pragma unroll
      for (int v = 0; v < 8; v++) {
        float x = acc[t][v] + bb[t];
        x = x > 0.f ? x : 0.f;
        runmax[t] = fmaxf(runmax[t], x);
      }
  }

  // combine the two 8-row halves within the wave, then across waves via LDS
#pragma unroll
  for (int t = 0; t < 4; t++)
    runmax[t] = fmaxf(runmax[t], __shfl_xor(runmax[t], 16, 32));
  if (hf == 0) {
#pragma unroll
    for (int t = 0; t < 4; t++) maxS[wave * 64 + t * 16 + r] = runmax[t];
  }
  __syncthreads();
  if (tid < 64) {
    float m = maxS[tid];
    m = fmaxf(m, maxS[64 + tid]);
    m = fmaxf(m, maxS[128 + tid]);
    m = fmaxf(m, maxS[192 + tid]);
    poolh[(size_t)i * 1024 + cb * 64 + tid] = (half_t)m;
  }
}

// ---------------------------------------------------------------------------
// Small scalar kernels
// ---------------------------------------------------------------------------
__global__ void embed2_kernel(const float* __restrict__ X,   // [M,2]
                              const float* __restrict__ W,   // [2,64]
                              const float* __restrict__ b,   // [64]
                              half_t* __restrict__ O, int M) {
  int idx = blockIdx.x * blockDim.x + threadIdx.x;
  if (idx >= M * 64) return;
  int m = idx >> 6, c = idx & 63;
  O[idx] = (half_t)(X[m * 2] * W[c] + X[m * 2 + 1] * W[64 + c] + b[c]);
}

__global__ void lstm_elem_kernel(const float* __restrict__ g,  // [128,4H]
                                 float* __restrict__ c,        // [128,H]
                                 half_t* __restrict__ h16, int hshift) {
  int idx = blockIdx.x * blockDim.x + threadIdx.x;
  const int H = 1 << hshift;
  if (idx >= 128 * H) return;
  int m = idx >> hshift, k = idx & (H - 1);
  const float* gm = g + ((size_t)m << (hshift + 2));
  float gi = gm[k], gf = gm[H + k], gg = gm[2 * H + k], go = gm[3 * H + k];
  float si = 1.f / (1.f + expf(-gi));
  float sf = 1.f / (1.f + expf(-gf));
  float so = 1.f / (1.f + expf(-go));
  float cc = sf * c[idx] + si * tanhf(gg);
  c[idx] = cc;
  h16[idx] = (half_t)(so * tanhf(cc));
}

// h2p + position update + output write + next-step input embedding, one block.
__global__ void h2p_embed_kernel(const half_t* __restrict__ h16,   // [128,128]
                                 const float* __restrict__ W,      // [128,2]
                                 const float* __restrict__ b,      // [2]
                                 const float* __restrict__ embW,   // [2,64]
                                 const float* __restrict__ embB,   // [64]
                                 float* __restrict__ curr_pos,     // [128,2] (+=)
                                 float* __restrict__ out,          // [128,2]
                                 half_t* __restrict__ x16) {       // [128,64]
  __shared__ float relS[256];
  int t = threadIdx.x;  // 256 threads
  int m = t >> 1, d = t & 1;
  float s = b[d];
  for (int k = 0; k < 128; k++) s += (float)h16[m * 128 + k] * W[k * 2 + d];
  relS[t] = s;
  out[t] = s;
  curr_pos[t] += s;
  __syncthreads();
  for (int u = t; u < 128 * 64; u += 256) {
    int mm = u >> 6, c = u & 63;
    x16[u] = (half_t)(relS[mm * 2] * embW[c] + relS[mm * 2 + 1] * embW[64 + c] +
                      embB[c]);
  }
}

__global__ void copy256_kernel(const float* __restrict__ s, float* __restrict__ d) {
  d[threadIdx.x] = s[threadIdx.x];
}

__global__ void zero_f16_kernel(half_t* p, int n) {
  int i = blockIdx.x * blockDim.x + threadIdx.x;
  if (i < n) p[i] = (half_t)0.f;
}
__global__ void zero_f32_kernel(float* p, int n) {
  int i = blockIdx.x * blockDim.x + threadIdx.x;
  if (i < n) p[i] = 0.f;
}

// ---------------------------------------------------------------------------
// Host orchestration
// ---------------------------------------------------------------------------
extern "C" void kernel_launch(void* const* d_in, const int* in_sizes, int n_in,
                              void* d_out, int out_size, void* d_ws, size_t ws_size,
                              hipStream_t stream) {
  (void)in_sizes; (void)n_in; (void)out_size; (void)ws_size;

  const float* obs_traj  = (const float*)d_in[0];
  const float* obs_rel   = (const float*)d_in[1];
  const float* enc_emb_W = (const float*)d_in[2];
  const float* enc_emb_b = (const float*)d_in[3];
  const float* enc_Wih   = (const float*)d_in[4];
  const float* enc_Whh   = (const float*)d_in[5];
  const float* enc_bih   = (const float*)d_in[6];
  const float* enc_bhh   = (const float*)d_in[7];
  const float* gp_emb_W  = (const float*)d_in[8];
  const float* gp_emb_b  = (const float*)d_in[9];
  const float* gp_W1     = (const float*)d_in[10];
  const float* gp_b1     = (const float*)d_in[11];
  const float* gp_W2     = (const float*)d_in[12];
  const float* gp_b2     = (const float*)d_in[13];
  const float* ctx_W1    = (const float*)d_in[14];
  const float* ctx_b1    = (const float*)d_in[15];
  const float* ctx_W2    = (const float*)d_in[16];
  const float* ctx_b2    = (const float*)d_in[17];
  const float* dec_emb_W = (const float*)d_in[18];
  const float* dec_emb_b = (const float*)d_in[19];
  const float* dec_Wih   = (const float*)d_in[20];
  const float* dec_Whh   = (const float*)d_in[21];
  const float* dec_bih   = (const float*)d_in[22];
  const float* dec_bhh   = (const float*)d_in[23];
  const float* h2p_W     = (const float*)d_in[24];
  const float* h2p_b     = (const float*)d_in[25];
  const float* dp_emb_W  = (const float*)d_in[26];
  const float* dp_emb_b  = (const float*)d_in[27];
  const float* dp_W1     = (const float*)d_in[28];
  const float* dp_b1     = (const float*)d_in[29];
  const float* dp_W2     = (const float*)d_in[30];
  const float* dp_b2     = (const float*)d_in[31];
  const float* dmlp_W1   = (const float*)d_in[32];
  const float* dmlp_b1   = (const float*)d_in[33];
  const float* dmlp_W2   = (const float*)d_in[34];
  const float* dmlp_b2   = (const float*)d_in[35];
  float* out = (float*)d_out;

  // ---- workspace carving ----
  size_t off = 0;
  auto carve = [&](size_t bytes) -> void* {
    void* p = (char*)d_ws + off;
    off += (bytes + 255) & ~(size_t)255;
    return p;
  };
  half_t* P_encWih  = (half_t*)carve((size_t)64 * 256 * 2);
  half_t* P_encWhh  = (half_t*)carve((size_t)64 * 256 * 2);
  half_t* P_gpW1    = (half_t*)carve((size_t)128 * 512 * 2);
  half_t* P_gpW2    = (half_t*)carve((size_t)512 * 1024 * 2);
  half_t* P_ctxW1   = (half_t*)carve((size_t)1088 * 1024 * 2);
  half_t* P_ctxW2   = (half_t*)carve((size_t)1024 * 128 * 2);
  half_t* P_decWih  = (half_t*)carve((size_t)64 * 512 * 2);
  half_t* P_decWhh  = (half_t*)carve((size_t)128 * 512 * 2);
  half_t* P_dpW1    = (half_t*)carve((size_t)192 * 512 * 2);
  half_t* P_dpW2    = (half_t*)carve((size_t)512 * 1024 * 2);
  half_t* P_dmlpW1  = (half_t*)carve((size_t)1152 * 1024 * 2);
  half_t* P_dmlpW2  = (half_t*)carve((size_t)1024 * 128 * 2);

  half_t* emb_obs16 = (half_t*)carve((size_t)1024 * 64 * 2);   // 8 x [128,64]
  half_t* enc_h16   = (half_t*)carve((size_t)128 * 64 * 2);
  float*  enc_c     = (float*)carve((size_t)128 * 64 * 4);
  half_t* dec_h16   = (half_t*)carve((size_t)128 * 128 * 2);
  float*  dec_c     = (float*)carve((size_t)128 * 128 * 4);
  float*  gates     = (float*)carve((size_t)128 * 512 * 4);
  float*  hW1       = (float*)carve((size_t)128 * 512 * 4);
  half_t* pre1      = (half_t*)carve((size_t)16384 * 512 * 2); // 16 MB
  half_t* poolh16   = (half_t*)carve((size_t)128 * 1024 * 2);
  half_t* t1_16     = (half_t*)carve((size_t)128 * 1024 * 2);
  half_t* x16       = (half_t*)carve((size_t)128 * 64 * 2);
  float*  curr_pos  = (float*)carve((size_t)256 * 4);

  auto pack = [&](const float* W, half_t* P, int K, int N) {
    int total = K * N;
    pack_b_kernel<<<(total + 255) / 256, 256, 0, stream>>>(W, P, K, N);
  };
  // dual-input GEMM launcher (A2 may be null)
  auto gemm2 = [&](const half_t* A1, int lda1, const half_t* Bp1, int KT1,
                   int ktS1, int ktC1, const half_t* A2, int lda2,
                   const half_t* Bp2, int KT2, int ktS2, int ktC2,
                   const float* b1, const float* b2, const float* rowBias,
                   int rbLd, float* C, half_t* O16, int N, int M, int flags) {
    dim3 g(N / 64, (M + 127) / 128);
    gemm16_kernel<<<g, 256, 0, stream>>>(A1, lda1, Bp1, KT1, ktS1, ktC1, A2, lda2,
                                         Bp2, KT2, ktS2, ktC2, b1, b2, rowBias,
                                         rbLd, C, O16, N, M, flags);
  };
  // pool(h16[128,hd], pos) -> poolh16[128,1024]
  auto pool = [&](const half_t* h16, int hLd, int hdTiles, const float* pos,
                  const float* eW, const float* eb, const half_t* PW1, int KT1,
                  const float* b1, const half_t* PW2, const float* b2) {
    // hW1 = h @ W1[E:, :]
    gemm2(h16, hLd, PW1, KT1, 2, hdTiles, nullptr, 0, nullptr, 0, 0, 0,
          nullptr, nullptr, nullptr, 0, hW1, nullptr, 512, 128, 0);
    // pre1 = relu(emb(rel) @ W1[:E, :] + b1 + hW1[row & 127]) -- emb in-register
    pool_l1_kernel<<<dim3(8, 128), 256, 0, stream>>>(pos, eW, eb, PW1, KT1, b1,
                                                     hW1, pre1);
    // poolh = max_j relu(pre1 @ W2 + b2)  (B async-staged in LDS)
    pool_l2max_kernel<<<dim3(16, 128), 128, 65536 + 1024, stream>>>(pre1, PW2, b2,
                                                                   poolh16);
  };

  // ---- pack all weights into WMMA B-fragment layout ----
  pack(enc_Wih, P_encWih, 64, 256);    pack(enc_Whh, P_encWhh, 64, 256);
  pack(gp_W1, P_gpW1, 128, 512);       pack(gp_W2, P_gpW2, 512, 1024);
  pack(ctx_W1, P_ctxW1, 1088, 1024);   pack(ctx_W2, P_ctxW2, 1024, 128);
  pack(dec_Wih, P_decWih, 64, 512);    pack(dec_Whh, P_decWhh, 128, 512);
  pack(dp_W1, P_dpW1, 192, 512);       pack(dp_W2, P_dpW2, 512, 1024);
  pack(dmlp_W1, P_dmlpW1, 1152, 1024); pack(dmlp_W2, P_dmlpW2, 1024, 128);

  // ---- encoder ----
  embed2_kernel<<<(1024 * 64) / 256, 256, 0, stream>>>(obs_rel, enc_emb_W,
                                                       enc_emb_b, emb_obs16, 1024);
  zero_f16_kernel<<<(128 * 64 + 255) / 256, 256, 0, stream>>>(enc_h16, 128 * 64);
  zero_f32_kernel<<<(128 * 64 + 255) / 256, 256, 0, stream>>>(enc_c, 128 * 64);
  for (int t = 0; t < 8; t++) {
    gemm2(emb_obs16 + (size_t)t * 128 * 64, 64, P_encWih, 2, 0, 2,
          enc_h16, 64, P_encWhh, 2, 0, 2, enc_bih, enc_bhh, nullptr, 0,
          gates, nullptr, 256, 128, 0);
    lstm_elem_kernel<<<(128 * 64) / 256, 256, 0, stream>>>(gates, enc_c, enc_h16, 6);
  }

  // ---- social pooling on encoder hidden (pos = obs_traj[-1]) ----
  copy256_kernel<<<1, 256, 0, stream>>>(obs_traj + 7 * 256, curr_pos);
  pool(enc_h16, 64, 2, curr_pos, gp_emb_W, gp_emb_b, P_gpW1, 4, gp_b1, P_gpW2, gp_b2);

  // ---- context MLP -> dec_h ----
  gemm2(enc_h16, 64, P_ctxW1, 34, 0, 2, poolh16, 1024, P_ctxW1, 34, 2, 32,
        ctx_b1, nullptr, nullptr, 0, nullptr, t1_16, 1024, 128, GF_RELU);
  gemm2(t1_16, 1024, P_ctxW2, 32, 0, 32, nullptr, 0, nullptr, 0, 0, 0,
        ctx_b2, nullptr, nullptr, 0, nullptr, dec_h16, 128, 128, GF_RELU);
  zero_f32_kernel<<<(128 * 128 + 255) / 256, 256, 0, stream>>>(dec_c, 128 * 128);

  // ---- decoder ----
  embed2_kernel<<<(128 * 64) / 256, 256, 0, stream>>>(obs_rel + 7 * 256, dec_emb_W,
                                                      dec_emb_b, x16, 128);
  for (int t = 0; t < 12; t++) {
    gemm2(x16, 64, P_decWih, 2, 0, 2, dec_h16, 128, P_decWhh, 4, 0, 4,
          dec_bih, dec_bhh, nullptr, 0, gates, nullptr, 512, 128, 0);
    lstm_elem_kernel<<<(128 * 128) / 256, 256, 0, stream>>>(gates, dec_c, dec_h16, 7);
    h2p_embed_kernel<<<1, 256, 0, stream>>>(dec_h16, h2p_W, h2p_b, dec_emb_W,
                                            dec_emb_b, curr_pos,
                                            out + (size_t)t * 256, x16);
    if (t < 11) {
      pool(dec_h16, 128, 4, curr_pos, dp_emb_W, dp_emb_b, P_dpW1, 6, dp_b1,
           P_dpW2, dp_b2);
      gemm2(dec_h16, 128, P_dmlpW1, 36, 0, 4, poolh16, 1024, P_dmlpW1, 36, 4, 32,
            dmlp_b1, nullptr, nullptr, 0, nullptr, t1_16, 1024, 128, GF_RELU);
      gemm2(t1_16, 1024, P_dmlpW2, 32, 0, 32, nullptr, 0, nullptr, 0, 0, 0,
            dmlp_b2, nullptr, nullptr, 0, nullptr, dec_h16, 128, 128, GF_RELU);
    }
  }
}
